// SimGRewGCN2Conv_29772713296409
// MI455X (gfx1250) — matrix operand order repaired and checked by reading, not compiled, not run on hardware
//
#include <hip/hip_runtime.h>

// ---------------------------------------------------------------------------
// GCNII forward on MI455X (gfx1250, wave32, WMMA).
//
// N=12288, F=256, H=64, C=16, L=4. Dominant cost: 5 passes over the 604MB
// dense adjacency (4 layer SpMMs + 1 Dirichlet pass) + 1 degree pass
// -> ~3.6GB HBM traffic ~ 160us @ 23.3TB/s. adj entries are exactly {0,1}
// (diag forced to exactly 1.0 after +I), so f16 WMMA on A is exact; the small
// B operand uses an f16 hi/lo split (two accumulators summed in fp32) for
// ~fp32-grade accuracy while staying HBM-bound (32 FLOP/byte on A << f16
// WMMA peak). Workspace use: ~13 MB. All reductions atomics-free =>
// bit-deterministic across graph replays.
// ---------------------------------------------------------------------------

#define Nn 12288
#define Ff 256
#define Hh 64
#define Cc 16
#define Ll 4

typedef __attribute__((ext_vector_type(16))) _Float16 v16h;
typedef __attribute__((ext_vector_type(8)))  _Float16 v8h;
typedef __attribute__((ext_vector_type(8)))  float    v8f;
typedef __attribute__((ext_vector_type(4)))  float    v4f;

// ---------------------------------------------------------------------------
// deg[u] = 1 + sum_v adj[u,v]   (A = adj + I);  dinv[u] = rsqrt(deg[u])
// ---------------------------------------------------------------------------
__global__ __launch_bounds__(256) void k_rowsum(const float* __restrict__ adj,
                                                float* __restrict__ deg,
                                                float* __restrict__ dinv) {
  const int wave = threadIdx.x >> 5, lane = threadIdx.x & 31;
  const int row = blockIdx.x * 8 + wave;
  const float* rp = adj + (size_t)row * Nn;
  float s = 0.f;
  for (int c = lane * 4; c < Nn; c += 128) {
    v4f v = *(const v4f*)(rp + c);
    s += v.x + v.y + v.z + v.w;
  }
  for (int o = 16; o >= 1; o >>= 1) s += __shfl_xor(s, o, 32);
  if (lane == 0) {
    float d = s + 1.0f;
    deg[row] = d;
    dinv[row] = rsqrtf(d);
  }
}

// ---------------------------------------------------------------------------
// h0 = x @ W_in + b_in  (tiny: 0.2 GFLOP; x row broadcast, W_in coalesced)
// ---------------------------------------------------------------------------
__global__ __launch_bounds__(256) void k_init_h(const float* __restrict__ x,
                                                const float* __restrict__ Win,
                                                const float* __restrict__ bin,
                                                float* __restrict__ h0) {
  const int idx = blockIdx.x * 256 + threadIdx.x;
  const int row = idx >> 6, col = idx & 63;
  const float* xr = x + (size_t)row * Ff;
  float acc = bin[col];
  for (int k = 0; k < Ff; ++k) acc += xr[k] * Win[k * Hh + col];
  h0[idx] = acc;
}

// ---------------------------------------------------------------------------
// M_i = (1-beta_i) I + beta_i * Wconv_i,  beta_i = log(0.3/(i+1) + 1)
// ---------------------------------------------------------------------------
__global__ __launch_bounds__(256) void k_make_M(const float* __restrict__ Wc,
                                                float* __restrict__ M) {
  const int idx = blockIdx.x * 256 + threadIdx.x;     // L*H*H = 16384
  const int i = idx >> 12;
  const int rc = idx & 4095;
  const int r = rc >> 6, c = rc & 63;
  const float beta = logf(0.3f / (float)(i + 1) + 1.0f);
  M[idx] = beta * Wc[idx] + ((r == c) ? (1.0f - beta) : 0.0f);
}

// ---------------------------------------------------------------------------
// B operand prep: y = dinv .* h, f16 hi/lo split.  y16[v][c]=hi, [v][64+c]=lo
// ---------------------------------------------------------------------------
__global__ __launch_bounds__(256) void k_make_y(const float* __restrict__ h,
                                                const float* __restrict__ dinv,
                                                _Float16* __restrict__ y16) {
  const int idx = blockIdx.x * 256 + threadIdx.x;     // N*H elements
  const int row = idx >> 6, col = idx & 63;
  const float v = dinv[row] * h[idx];
  const _Float16 hi = (_Float16)v;
  y16[(size_t)row * 128 + col] = hi;
  y16[(size_t)row * 128 + 64 + col] = (_Float16)(v - (float)hi);
}

// ---------------------------------------------------------------------------
// Core WMMA SpMM:  out[N][L] = dinv_row * ( (adj + I) @ B )
//   B: [N][NC] f16, NC = 2*L (hi cols 0..L-1, lo cols L..NC-1).
// 256 threads (8 waves), 128 output rows per block, full-K loop (384 iters).
// A tile (128x32 fp32) streamed with non-temporal loads and converted to f16
// in LDS; B (<=3MB) stays L2-hot. The diagonal of A' is exactly 1.0 (adj has
// a zeroed diagonal), written once per thread at its precomputed (k0, jd).
// Fragment layouts follow the CDNA5 ISA 16-bit A(16x32)/B(32x16) tables.
// ---------------------------------------------------------------------------
template <int NC>
__global__ __launch_bounds__(256) void k_spmm(const float* __restrict__ adj,
                                              const _Float16* __restrict__ B,
                                              const float* __restrict__ dinv,
                                              float* __restrict__ out) {
  constexpr int NT  = NC / 16;   // WMMA tiles per wave per k-step
  constexpr int L   = NC / 2;    // logical output columns
  constexpr int NTL = NT / 2;
  constexpr int TPK = NC / 8;    // threads covering one k-row of B
  constexpr int KN  = NC / 64;   // k-rows of B per thread

  __shared__ _Float16 ldsA[128][40];   // +8 halves pad: bank-spread, 16B aligned
  __shared__ _Float16 ldsB[NC][40];    // stored transposed: ldsB[n][k]

  const int tid  = threadIdx.x;
  const int wave = tid >> 5;
  const int lane = tid & 31;
  const int R0   = blockIdx.x * 128;

  // A-tile ownership: 2 threads per row, 16 columns each
  const int arow = tid >> 1;
  const int acol = (tid & 1) * 16;
  const int gr   = R0 + arow;
  const float* aptr = adj + (size_t)gr * Nn + acol;

  // Diagonal crossing for this thread: happens at exactly one k-step.
  // Window at step k0 covers global cols [k0+acol, k0+acol+16).
  const int dpos = gr - acol;
  const bool hasd = (dpos >= 0) && ((dpos & 31) < 16);
  const int k0d = dpos & ~31;
  const int jd  = dpos & 31;     // < 16 when hasd

  // B-tile ownership: v8h rows of B, scattered into transposed LDS
  const int bk  = tid / TPK;
  const int bn0 = (tid % TPK) * 8;

  // fragment geometry (ISA 7.12.2 layouts, wave32)
  const int frow = wave * 16 + (lane & 15);
  const int kbA  = (lane >> 4) * 8;    // A: lanes<16 -> K{0..7,16..23}
  const int kbB  = (lane >> 4) * 8;    // B: lanes<16 -> K{0..7,16..23}
  const int j16  = lane & 15;

  v8f acc[NT];
#pragma unroll
  for (int t = 0; t < NT; ++t)
#pragma unroll
    for (int q = 0; q < 8; ++q) acc[t][q] = 0.0f;

  v4f fa[4];
  v8h fb[KN];
  auto load_tile = [&](int k0) {
#pragma unroll
    for (int q = 0; q < 4; ++q)
      fa[q] = __builtin_nontemporal_load((const v4f*)(aptr + k0 + q * 4));
#pragma unroll
    for (int t = 0; t < KN; ++t) {
      const int kk = bk + t * (256 / TPK);
      fb[t] = *(const v8h*)(B + (size_t)(k0 + kk) * NC + bn0);
    }
  };

  load_tile(0);

  for (int k0 = 0; k0 < Nn; k0 += 32) {
    __syncthreads();   // previous k-step finished reading LDS
    // ---- stage A: fp32 -> f16 ----
    {
      v8h alo, ahi;
      alo[0] = (_Float16)fa[0].x; alo[1] = (_Float16)fa[0].y;
      alo[2] = (_Float16)fa[0].z; alo[3] = (_Float16)fa[0].w;
      alo[4] = (_Float16)fa[1].x; alo[5] = (_Float16)fa[1].y;
      alo[6] = (_Float16)fa[1].z; alo[7] = (_Float16)fa[1].w;
      ahi[0] = (_Float16)fa[2].x; ahi[1] = (_Float16)fa[2].y;
      ahi[2] = (_Float16)fa[2].z; ahi[3] = (_Float16)fa[2].w;
      ahi[4] = (_Float16)fa[3].x; ahi[5] = (_Float16)fa[3].y;
      ahi[6] = (_Float16)fa[3].z; ahi[7] = (_Float16)fa[3].w;
      *(v8h*)&ldsA[arow][acol]     = alo;
      *(v8h*)&ldsA[arow][acol + 8] = ahi;
      // A + I: diagonal entry is exactly 1.0 (adj diag is zeroed upstream)
      if (hasd && k0 == k0d) ldsA[arow][acol + jd] = (_Float16)1.0f;
    }
    // ---- stage B transposed ----
#pragma unroll
    for (int t = 0; t < KN; ++t) {
      const int kk = bk + t * (256 / TPK);
#pragma unroll
      for (int q = 0; q < 8; ++q) ldsB[bn0 + q][kk] = fb[t][q];
    }
    __syncthreads();
    if (k0 + 32 < Nn) load_tile(k0 + 32);   // overlap next loads with WMMA

    // ---- build all fragments first (batch the DS loads), then WMMA chain ----
    v16h afr;
    {
      v8h lo = *(const v8h*)&ldsA[frow][kbA];
      v8h hi = *(const v8h*)&ldsA[frow][kbA + 16];
#pragma unroll
      for (int q = 0; q < 8; ++q) { afr[q] = lo[q]; afr[q + 8] = hi[q]; }
    }
    v16h bfr[NT];
#pragma unroll
    for (int cg = 0; cg < NT; ++cg) {
      const int bn = cg * 16 + j16;
      v8h b0 = *(const v8h*)&ldsB[bn][kbB];
      v8h b1 = *(const v8h*)&ldsB[bn][kbB + 16];
#pragma unroll
      for (int q = 0; q < 8; ++q) { bfr[cg][q] = b0[q]; bfr[cg][q + 8] = b1[q]; }
    }
#pragma unroll
    for (int cg = 0; cg < NT; ++cg) {
      acc[cg] = __builtin_amdgcn_wmma_f32_16x16x32_f16(
          false, afr, false, bfr[cg], (short)0, acc[cg], false, false);
    }
  }

  // ---- epilogue: D = dinv_row * (acc_hi + acc_lo) ----
  const int rbase = R0 + wave * 16 + (lane >> 4) * 8;
#pragma unroll
  for (int cg = 0; cg < NTL; ++cg) {
    const int col = cg * 16 + j16;
#pragma unroll
    for (int e = 0; e < 8; ++e) {
      const int r = rbase + e;
      out[(size_t)r * L + col] = dinv[r] * (acc[cg][e] + acc[cg + NTL][e]);
    }
  }
}

// ---------------------------------------------------------------------------
// Layer epilogue: hm = 0.7*agg + 0.3*h0; t = hm @ M_i; (ReLU + LayerNorm
// except last layer). One wave per row (lane covers cols j, j+32).
// ---------------------------------------------------------------------------
__global__ __launch_bounds__(256) void k_layer_post(const float* __restrict__ agg,
                                                    const float* __restrict__ h0,
                                                    const float* __restrict__ M,
                                                    const float* __restrict__ lng,
                                                    const float* __restrict__ lnb,
                                                    float* __restrict__ h,
                                                    int last) {
  __shared__ float ldsR[8][64];
  const int wave = threadIdx.x >> 5, lane = threadIdx.x & 31;
  const int row = blockIdx.x * 8 + wave;
  const size_t base = (size_t)row * Hh;
  const float hm0 = 0.7f * agg[base + lane]      + 0.3f * h0[base + lane];
  const float hm1 = 0.7f * agg[base + lane + 32] + 0.3f * h0[base + lane + 32];
  ldsR[wave][lane] = hm0;
  ldsR[wave][lane + 32] = hm1;
  __syncthreads();
  float t0 = 0.f, t1 = 0.f;
#pragma unroll 8
  for (int k = 0; k < 64; ++k) {
    const float hv = ldsR[wave][k];
    t0 += hv * M[k * 64 + lane];
    t1 += hv * M[k * 64 + lane + 32];
  }
  if (!last) {
    t0 = fmaxf(t0, 0.f);
    t1 = fmaxf(t1, 0.f);
    float s = t0 + t1;
    for (int o = 16; o >= 1; o >>= 1) s += __shfl_xor(s, o, 32);
    const float mu = s * (1.0f / 64.0f);
    const float d0 = t0 - mu, d1 = t1 - mu;
    float v = d0 * d0 + d1 * d1;
    for (int o = 16; o >= 1; o >>= 1) v += __shfl_xor(v, o, 32);
    const float rs = rsqrtf(v * (1.0f / 64.0f) + 1e-5f);
    t0 = d0 * rs * lng[lane]      + lnb[lane];
    t1 = d1 * rs * lng[lane + 32] + lnb[lane + 32];
  }
  h[base + lane] = t0;
  h[base + lane + 32] = t1;
}

// ---------------------------------------------------------------------------
// out = h @ W_out + b_out; logp = log_softmax(out); g = relu(out)/sqrt(deg+1);
// s = rowsum(g^2); energy-pass B operand yE[N][64] (f16 hi/lo, 32 logical
// cols: 0..15 = dinv*g, 16 = dinv (yields W rowsum), 17..31 = 0).
// ---------------------------------------------------------------------------
__global__ __launch_bounds__(256) void k_out(const float* __restrict__ h,
                                             const float* __restrict__ Wo,
                                             const float* __restrict__ bo,
                                             const float* __restrict__ deg,
                                             const float* __restrict__ dinv,
                                             float* __restrict__ outbuf,
                                             float* __restrict__ svec,
                                             _Float16* __restrict__ yE) {
  const int wave = threadIdx.x >> 5, lane = threadIdx.x & 31;
  const int half = lane >> 4, j = lane & 15;
  const int row = blockIdx.x * 16 + wave * 2 + half;
  const size_t hb = (size_t)row * Hh;
  float o = bo[j];
  for (int k = 0; k < 64; ++k) o += h[hb + k] * Wo[k * Cc + j];
  outbuf[(size_t)row * Cc + j] = o;
  // log-softmax across the 16-lane group (xor masks 1..8 stay inside group)
  float m = o;
  for (int t = 8; t >= 1; t >>= 1) m = fmaxf(m, __shfl_xor(m, t, 32));
  const float l = o - m;
  float se = expf(l);
  for (int t = 8; t >= 1; t >>= 1) se += __shfl_xor(se, t, 32);
  outbuf[(size_t)Nn * Cc + (size_t)row * Cc + j] = l - logf(se);
  // Dirichlet pieces
  const float gv = fmaxf(o, 0.f) * rsqrtf(deg[row] + 1.0f);
  float s2 = gv * gv;
  for (int t = 8; t >= 1; t >>= 1) s2 += __shfl_xor(s2, t, 32);
  if (j == 0) svec[row] = s2;
  const float dv = dinv[row];
  const float gd = gv * dv;
  const _Float16 hi = (_Float16)gd;
  yE[(size_t)row * 64 + j]      = hi;
  yE[(size_t)row * 64 + 32 + j] = (_Float16)(gd - (float)hi);
  if (j == 0) {
    const _Float16 dh = (_Float16)dv;
    yE[(size_t)row * 64 + 16] = dh;
    yE[(size_t)row * 64 + 48] = (_Float16)(dv - (float)dh);
  } else {
    yE[(size_t)row * 64 + 16 + j] = (_Float16)0.f;
    yE[(size_t)row * 64 + 48 + j] = (_Float16)0.f;
  }
}

// ---------------------------------------------------------------------------
// Deterministic two-stage reduction for the Dirichlet energy (no atomics).
// E[N][32]: cols 0..15 = Wc@g, col 16 = rW.
// de = (sum rW*s - sum (Wc@g).g) / sum rW   (W symmetric).
// ---------------------------------------------------------------------------
__global__ __launch_bounds__(256) void k_energy_part(const float* __restrict__ E,
                                                     const float* __restrict__ svec,
                                                     const float* __restrict__ outbuf,
                                                     const float* __restrict__ deg,
                                                     float* __restrict__ partials) {
  __shared__ float red[3][256];
  const int tid = threadIdx.x;
  const int row = blockIdx.x * 256 + tid;
  const float rW = E[(size_t)row * 32 + 16];
  const float ir = rsqrtf(deg[row] + 1.0f);
  float p1 = 0.f;
#pragma unroll
  for (int j = 0; j < 16; ++j) {
    const float gv = fmaxf(outbuf[(size_t)row * Cc + j], 0.f) * ir;
    p1 += E[(size_t)row * 32 + j] * gv;
  }
  red[0][tid] = rW * svec[row];
  red[1][tid] = p1;
  red[2][tid] = rW;
  __syncthreads();
  for (int s = 128; s >= 1; s >>= 1) {
    if (tid < s) {
      red[0][tid] += red[0][tid + s];
      red[1][tid] += red[1][tid + s];
      red[2][tid] += red[2][tid + s];
    }
    __syncthreads();
  }
  if (tid == 0) {
    partials[blockIdx.x * 3 + 0] = red[0][0];
    partials[blockIdx.x * 3 + 1] = red[1][0];
    partials[blockIdx.x * 3 + 2] = red[2][0];
  }
}

__global__ void k_energy_final(const float* __restrict__ partials,
                               float* __restrict__ de_out) {
  if (threadIdx.x == 0) {
    float p0 = 0.f, p1 = 0.f, p2 = 0.f;
    for (int i = 0; i < 48; ++i) {
      p0 += partials[i * 3 + 0];
      p1 += partials[i * 3 + 1];
      p2 += partials[i * 3 + 2];
    }
    de_out[0] = (p0 - p1) / p2;
  }
}

// ---------------------------------------------------------------------------
extern "C" void kernel_launch(void* const* d_in, const int* in_sizes, int n_in,
                              void* d_out, int out_size, void* d_ws, size_t ws_size,
                              hipStream_t stream) {
  (void)in_sizes; (void)n_in; (void)out_size; (void)ws_size;
  const float* x     = (const float*)d_in[0];
  const float* adj   = (const float*)d_in[1];
  const float* W_in  = (const float*)d_in[2];
  const float* b_in  = (const float*)d_in[3];
  const float* W_cv  = (const float*)d_in[4];
  const float* ln_g  = (const float*)d_in[5];
  const float* ln_b  = (const float*)d_in[6];
  const float* W_out = (const float*)d_in[7];
  const float* b_out = (const float*)d_in[8];
  float* out = (float*)d_out;

  // workspace carve-up (~13 MB, 256B-aligned offsets)
  char* w = (char*)d_ws;
  const size_t SZ_VEC = (size_t)Nn * 4;            // 48 KB
  const size_t SZ_MAT = (size_t)Nn * Hh * 4;       // 3 MB
  float*    deg   = (float*)(w);
  float*    dinv  = (float*)(w + SZ_VEC);
  float*    h0    = (float*)(w + 2 * SZ_VEC);
  float*    h     = (float*)(w + 2 * SZ_VEC + SZ_MAT);
  float*    agg   = (float*)(w + 2 * SZ_VEC + 2 * SZ_MAT);   // also E[N][32]
  _Float16* Bbuf  = (_Float16*)(w + 2 * SZ_VEC + 3 * SZ_MAT);           // [N][128] f16
  float*    Mbuf  = (float*)(w + 2 * SZ_VEC + 3 * SZ_MAT + (size_t)Nn * 128 * 2);
  float*    svec  = (float*)((char*)Mbuf + (size_t)Ll * Hh * Hh * 4);
  float*    parts = (float*)((char*)svec + SZ_VEC);

  dim3 blk(256);
  k_rowsum<<<dim3(Nn / 8), blk, 0, stream>>>(adj, deg, dinv);
  k_init_h<<<dim3(Nn * Hh / 256), blk, 0, stream>>>(x, W_in, b_in, h0);
  k_make_M<<<dim3(Ll * Hh * Hh / 256), blk, 0, stream>>>(W_cv, Mbuf);

  for (int i = 0; i < Ll; ++i) {
    const float* cur = (i == 0) ? h0 : h;
    k_make_y<<<dim3(Nn * Hh / 256), blk, 0, stream>>>(cur, dinv, Bbuf);
    k_spmm<128><<<dim3(Nn / 128), blk, 0, stream>>>(adj, Bbuf, dinv, agg);
    k_layer_post<<<dim3(Nn / 8), blk, 0, stream>>>(agg, h0, Mbuf + (size_t)i * Hh * Hh,
                                                   ln_g + i * Hh, ln_b + i * Hh,
                                                   h, (i == Ll - 1) ? 1 : 0);
  }

  k_out<<<dim3(Nn / 16), blk, 0, stream>>>(h, W_out, b_out, deg, dinv, out, svec, Bbuf);
  k_spmm<64><<<dim3(Nn / 128), blk, 0, stream>>>(adj, Bbuf, dinv, agg);
  k_energy_part<<<dim3(Nn / 256), blk, 0, stream>>>(agg, svec, out, deg, parts);
  k_energy_final<<<dim3(1), dim3(64), 0, stream>>>(parts, out + 2 * (size_t)Nn * Cc);
}
